// GPT_26465588478377
// MI455X (gfx1250) — matrix operand order repaired and checked by reading, not compiled, not run on hardware
//
#include <hip/hip_runtime.h>
#include <math.h>

typedef __bf16 bf16;
typedef __attribute__((ext_vector_type(16))) __bf16 v16bf;
typedef __attribute__((ext_vector_type(8)))  float  v8f;

union FragBF { v16bf v; unsigned u[8]; };
union FragF  { v8f   v; float    f[8]; };
union Pack8  { uint4 q; bf16 h[8]; };

#define GPT_T   2048
#define GPT_D   768
#define GPT_3D  2304
#define GPT_4D  3072
#define GPT_M   4096      // B*T
#define GPT_V   50257
#define GPT_L   4

// K index inside a 16x32 bf16 A-fragment for vgpr-pair i (0..7), half = lane>>4.
__device__ __forceinline__ int a_kbase(int i, int half) {
    return (i < 4) ? (i * 2 + half * 8) : (16 + (i - 4) * 2 + half * 8);
}

// ---- CDNA5 async global->LDS copy (ASYNCcnt-tracked, bypasses VGPRs) --------
// Flat addresses of __shared__ objects carry the LDS byte offset in the low 32
// bits (ISA: LDS aperture).
__device__ __forceinline__ unsigned lds_addr(const void* p) {
    return (unsigned)(uintptr_t)p;
}
__device__ __forceinline__ void async_copy_b128(unsigned loff, const void* g) {
    asm volatile("global_load_async_to_lds_b128 %0, %1, off"
                 :: "v"(loff), "v"(g) : "memory");
}
__device__ __forceinline__ void wait_async0() {
    asm volatile("s_wait_asynccnt 0x0" ::: "memory");
}

// ------------------------------------------------- fp32 [L,K,N] -> bf16 [L,N,K]
// 64x64 tile transpose through LDS; one-time per launch.
__global__ __launch_bounds__(256) void cvt_transpose_kernel(const float* __restrict__ in,
                                                            bf16* __restrict__ out,
                                                            int K, int N) {
    __shared__ bf16 tile[64][72];       // padded rows vs bank conflicts
    const int l = blockIdx.z;
    const float* inl = in + (size_t)l * K * N;
    bf16* outl = out + (size_t)l * K * N;
    const int k0 = blockIdx.y * 64, n0 = blockIdx.x * 64;
    const int tid = threadIdx.x;

    const int tr = tid >> 4;            // 0..15
    const int tc = (tid & 15) * 4;      // 0..60
    #pragma unroll
    for (int rr = 0; rr < 4; ++rr) {
        int row = tr + rr * 16;
        float4 f = *reinterpret_cast<const float4*>(inl + (size_t)(k0 + row) * N + n0 + tc);
        tile[row][tc + 0] = (bf16)f.x;
        tile[row][tc + 1] = (bf16)f.y;
        tile[row][tc + 2] = (bf16)f.z;
        tile[row][tc + 3] = (bf16)f.w;
    }
    __syncthreads();
    const int n = tid >> 2;             // 0..63
    const int kq = (tid & 3) * 16;      // 0,16,32,48
    Pack8 p0, p1;
    #pragma unroll
    for (int i = 0; i < 8; ++i) { p0.h[i] = tile[kq + i][n]; p1.h[i] = tile[kq + 8 + i][n]; }
    bf16* orow = outl + (size_t)(n0 + n) * K + k0 + kq;
    *reinterpret_cast<uint4*>(orow)     = p0.q;
    *reinterpret_cast<uint4*>(orow + 8) = p1.q;
}

// ---------------------------------------------------------------- embedding
__global__ __launch_bounds__(256) void embed_kernel(const int* __restrict__ idx,
                                                    const float* __restrict__ wte,
                                                    const float* __restrict__ wpe,
                                                    float* __restrict__ X) {
    int row = blockIdx.x;                  // 0..4095  (b*T + t)
    int tok = idx[row];
    int t = row & (GPT_T - 1);
    const float* wr = wte + (size_t)tok * GPT_D;
    const float* pr = wpe + (size_t)t * GPT_D;
    float* xr = X + (size_t)row * GPT_D;
    for (int i = threadIdx.x; i < GPT_D; i += 256) xr[i] = wr[i] + pr[i];
}

// ---------------------------------------------------------------- layernorm -> bf16
__global__ __launch_bounds__(256) void ln_kernel(const float* __restrict__ x,
                                                 const float* __restrict__ w,
                                                 const float* __restrict__ bb,
                                                 bf16* __restrict__ out) {
    int lane = threadIdx.x & 31;
    int row = blockIdx.x * 8 + (threadIdx.x >> 5);
    const float* xr = x + (size_t)row * GPT_D;
    float s = 0.f;
    for (int i = lane; i < GPT_D; i += 32) s += xr[i];
    #pragma unroll
    for (int o = 16; o > 0; o >>= 1) s += __shfl_xor(s, o, 32);
    float mu = s * (1.0f / GPT_D);
    float v = 0.f;
    for (int i = lane; i < GPT_D; i += 32) { float d = xr[i] - mu; v += d * d; }
    #pragma unroll
    for (int o = 16; o > 0; o >>= 1) v += __shfl_xor(v, o, 32);
    float rs = rsqrtf(v * (1.0f / GPT_D) + 1e-5f);
    bf16* orow = out + (size_t)row * GPT_D;
    for (int i = lane; i < GPT_D; i += 32)
        orow[i] = (bf16)((xr[i] - mu) * rs * w[i] + bb[i]);
}

// final LN, only last token of each batch, fp32 out [2][768]
__global__ void lnf_kernel(const float* __restrict__ x,
                           const float* __restrict__ w,
                           const float* __restrict__ bb,
                           float* __restrict__ out) {
    int lane = threadIdx.x & 31;
    int wv = threadIdx.x >> 5;             // 0,1  -> batch
    int row = (wv == 0) ? (GPT_T - 1) : (2 * GPT_T - 1);
    const float* xr = x + (size_t)row * GPT_D;
    float s = 0.f;
    for (int i = lane; i < GPT_D; i += 32) s += xr[i];
    #pragma unroll
    for (int o = 16; o > 0; o >>= 1) s += __shfl_xor(s, o, 32);
    float mu = s * (1.0f / GPT_D);
    float v = 0.f;
    for (int i = lane; i < GPT_D; i += 32) { float d = xr[i] - mu; v += d * d; }
    #pragma unroll
    for (int o = 16; o > 0; o >>= 1) v += __shfl_xor(v, o, 32);
    float rs = rsqrtf(v * (1.0f / GPT_D) + 1e-5f);
    for (int i = lane; i < GPT_D; i += 32)
        out[wv * GPT_D + i] = (xr[i] - mu) * rs * w[i] + bb[i];
}

// ---------------------------------------------------------------- WMMA GEMM
// C[M,N] = act( A_bf16[M,K] @ Wt_bf16[N,K]^T + bias ) (+ resid)
// block: 256 thr = 8 waves, 128x64 tile; wave(wm,wn): 64x16 (4 WMMA tiles).
// double-buffered LDS; tiles staged by the CDNA5 async global->LDS engine;
// per k-step: wait_asynccnt(0) -> barrier -> issue next tile -> 4x WMMA.
__global__ __launch_bounds__(256) void gemm_kernel(const bf16* __restrict__ A,
                                                   const bf16* __restrict__ Wt,
                                                   const float* __restrict__ bias,
                                                   const float* __restrict__ resid,
                                                   float* __restrict__ outF,
                                                   bf16* __restrict__ outB,
                                                   int M, int N, int K, int gelu) {
    __shared__ bf16 As[2][128][32];   // [buf][m][k]
    __shared__ bf16 Ws[2][64][32];    // [buf][n][k]

    const int tid = threadIdx.x;
    const int lane = tid & 31, wid = tid >> 5;
    const int half = lane >> 4, l16 = lane & 15;
    const int wm = wid >> 2, wn = wid & 3;          // wm 0..1 (64 rows), wn 0..3
    const int bM = blockIdx.y * 128, bN = blockIdx.x * 64;

    FragF acc[4];
    #pragma unroll
    for (int mt = 0; mt < 4; ++mt)
        #pragma unroll
        for (int r = 0; r < 8; ++r) acc[mt].f[r] = 0.f;

    const int am = tid >> 1;              // 0..127
    const int ak = (tid & 1) * 16;        // 0,16
    const int wn_ = tid >> 2;             // 0..63
    const int wk0 = (tid & 3) * 8;        // 0,8,16,24

    const bf16* aptr = A  + (size_t)(bM + am) * K + ak;
    const bf16* wptr = Wt + (size_t)(bN + wn_) * K + wk0;

    const unsigned la0[2] = { lds_addr(&As[0][am][ak]),     lds_addr(&As[1][am][ak]) };
    const unsigned la1[2] = { lds_addr(&As[0][am][ak + 8]), lds_addr(&As[1][am][ak + 8]) };
    const unsigned lw[2]  = { lds_addr(&Ws[0][wn_][wk0]),   lds_addr(&Ws[1][wn_][wk0]) };

    // prologue: async-stage tile 0
    async_copy_b128(la0[0], aptr);
    async_copy_b128(la1[0], aptr + 8);
    async_copy_b128(lw[0],  wptr);

    const int nk = K >> 5;
    for (int j = 0; j < nk; ++j) {
        wait_async0();                 // my async loads for tile j have landed
        __syncthreads();               // ... and everyone else's too
        const int cur = j & 1;
        if (j + 1 < nk) {              // DMA next tile into the other buffer
            const int kt = (j + 1) * 32;
            const int nxt = cur ^ 1;
            async_copy_b128(la0[nxt], aptr + kt);
            async_copy_b128(la1[nxt], aptr + kt + 8);
            async_copy_b128(lw[nxt],  wptr + kt);
            if (j + 2 < nk) {          // L2 prefetch of the k+2 tile
                __builtin_prefetch(aptr + kt + 32, 0, 1);
                __builtin_prefetch(wptr + kt + 32, 0, 1);
            }
        }
        // compute from current buffer
        FragBF bf_;
        #pragma unroll
        for (int i = 0; i < 8; ++i)
            bf_.u[i] = *reinterpret_cast<const unsigned*>(
                &Ws[cur][wn * 16 + l16][a_kbase(i, half)]);
        #pragma unroll
        for (int mt = 0; mt < 4; ++mt) {
            FragBF af;
            const bf16* arow = &As[cur][wm * 64 + mt * 16 + l16][0];
            #pragma unroll
            for (int i = 0; i < 8; ++i)
                af.u[i] = *reinterpret_cast<const unsigned*>(arow + a_kbase(i, half));
            acc[mt].v = __builtin_amdgcn_wmma_f32_16x16x32_bf16(
                false, af.v, false, bf_.v, (short)0, acc[mt].v, false, false);
        }
    }

    #pragma unroll
    for (int mt = 0; mt < 4; ++mt)
        #pragma unroll
        for (int r = 0; r < 8; ++r) {
            int gm = bM + wm * 64 + mt * 16 + r + 8 * half;
            int gn = bN + wn * 16 + l16;
            float v = acc[mt].f[r] + bias[gn];
            if (gelu) v = 0.5f * v * (1.0f + erff(v * 0.70710678118654752f));
            if (resid) v += resid[(size_t)gm * N + gn];
            if (outF) outF[(size_t)gm * N + gn] = v;
            if (outB) outB[(size_t)gm * N + gn] = (bf16)v;
        }
}

// ---------------------------------------------------------------- flash attention
// block = 256 thr = 8 waves; each block: one (b,h), 128 q rows; wave: 16 q rows.
// K tile staged by async global->LDS; V tile register-staged (transposed).
__global__ __launch_bounds__(256) void attn_kernel(const bf16* __restrict__ qkv,
                                                   bf16* __restrict__ y) {
    __shared__ bf16 Kt[2][32][64];      // [buf][key][hd]
    __shared__ bf16 VtT[2][64][32];     // [buf][hd][key]
    __shared__ bf16 Pms[8][16][32];     // per-wave P relayout scratch

    const int tid = threadIdx.x;
    const int lane = tid & 31, wid = tid >> 5;
    const int half = lane >> 4, l16 = lane & 15;
    const int qb = blockIdx.x;          // 0..15
    const int h = blockIdx.y;           // 0..11
    const int b = blockIdx.z;           // 0..1
    const int qr0 = qb * 128 + wid * 16;

    const bf16* base = qkv + (size_t)(b * GPT_T) * GPT_3D;

    // Q fragments (A-matrix 16x64 = two 16x32 frags), direct from global
    FragBF qf[2];
    {
        const bf16* qrow = base + (size_t)(qr0 + l16) * GPT_3D + h * 64;
        #pragma unroll
        for (int f = 0; f < 2; ++f)
            #pragma unroll
            for (int i = 0; i < 8; ++i)
                qf[f].u[i] = *reinterpret_cast<const unsigned*>(qrow + f * 32 + a_kbase(i, half));
    }

    FragF o[4];
    #pragma unroll
    for (int s = 0; s < 4; ++s)
        #pragma unroll
        for (int r = 0; r < 8; ++r) o[s].f[r] = 0.f;
    float m_run[8], l_run[8];
    #pragma unroll
    for (int r = 0; r < 8; ++r) { m_run[r] = -1e30f; l_run[r] = 0.f; }

    const int ntiles = (qb + 1) * 4;              // 32-key tiles up to causal edge
    const int kk = tid >> 3;                      // 0..31
    const int d0 = (tid & 7) * 8;                 // 0..56

    const bf16* krow0 = base + (size_t)kk * GPT_3D + GPT_D + h * 64 + d0;
    const size_t kstep = (size_t)32 * GPT_3D;     // 32 keys

    const unsigned lk[2] = { lds_addr(&Kt[0][kk][d0]), lds_addr(&Kt[1][kk][d0]) };

    Pack8 vv;
    async_copy_b128(lk[0], krow0);                           // K tile 0 via DMA
    vv.q = *reinterpret_cast<const uint4*>(krow0 + GPT_D);   // V tile 0 via regs
    #pragma unroll
    for (int i = 0; i < 8; ++i) VtT[0][d0 + i][kk] = vv.h[i];

    for (int j = 0; j < ntiles; ++j) {
        wait_async0();
        __syncthreads();
        const int cur = j & 1;
        const int k0 = j * 32;
        const bool more = (j + 1) < ntiles;
        if (more) {
            const bf16* krow = krow0 + (size_t)(j + 1) * kstep;
            async_copy_b128(lk[cur ^ 1], krow);
            vv.q = *reinterpret_cast<const uint4*>(krow + GPT_D);
        }

        if (k0 <= qr0 + 15) {                     // wave-uniform -> EXEC all ones
            // S = Q @ K^T   (two 16x16 key tiles, contraction over hd=64)
            FragF s[2];
            #pragma unroll
            for (int t = 0; t < 2; ++t)
                #pragma unroll
                for (int r = 0; r < 8; ++r) s[t].f[r] = 0.f;
            #pragma unroll
            for (int t = 0; t < 2; ++t)
                #pragma unroll
                for (int ks = 0; ks < 2; ++ks) {
                    FragBF bk;
                    #pragma unroll
                    for (int i = 0; i < 8; ++i)
                        bk.u[i] = *reinterpret_cast<const unsigned*>(
                            &Kt[cur][t * 16 + l16][ks * 32 + a_kbase(i, half)]);
                    s[t].v = __builtin_amdgcn_wmma_f32_16x16x32_bf16(
                        false, qf[ks].v, false, bk.v, (short)0, s[t].v, false, false);
                }
            // scale + causal mask
            #pragma unroll
            for (int t = 0; t < 2; ++t)
                #pragma unroll
                for (int r = 0; r < 8; ++r) {
                    int key = k0 + t * 16 + l16;
                    int qq = qr0 + r + 8 * half;
                    s[t].f[r] = (key <= qq) ? s[t].f[r] * 0.125f : -1e30f;
                }
            // online softmax per row
            #pragma unroll
            for (int r = 0; r < 8; ++r) {
                float mx = fmaxf(s[0].f[r], s[1].f[r]);
                #pragma unroll
                for (int off = 8; off > 0; off >>= 1) mx = fmaxf(mx, __shfl_xor(mx, off, 32));
                float mn = fmaxf(m_run[r], mx);
                float alpha = expf(m_run[r] - mn);
                float p0 = expf(s[0].f[r] - mn);
                float p1 = expf(s[1].f[r] - mn);
                s[0].f[r] = p0; s[1].f[r] = p1;
                float rs = p0 + p1;
                #pragma unroll
                for (int off = 8; off > 0; off >>= 1) rs += __shfl_xor(rs, off, 32);
                l_run[r] = l_run[r] * alpha + rs;
                m_run[r] = mn;
                #pragma unroll
                for (int ss = 0; ss < 4; ++ss) o[ss].f[r] *= alpha;
            }
            // relayout P (C-frag -> A-frag) through wave-private LDS
            bf16* Pm = &Pms[wid][0][0];
            #pragma unroll
            for (int t = 0; t < 2; ++t)
                #pragma unroll
                for (int r = 0; r < 8; ++r)
                    Pm[(r + 8 * half) * 32 + t * 16 + l16] = (bf16)s[t].f[r];
            FragBF pf;
            {
                const bf16* prow = Pm + l16 * 32;
                #pragma unroll
                for (int i = 0; i < 8; ++i)
                    pf.u[i] = *reinterpret_cast<const unsigned*>(prow + a_kbase(i, half));
            }
            // O += P @ V   (4 hd slices)
            #pragma unroll
            for (int ss = 0; ss < 4; ++ss) {
                FragBF bv;
                #pragma unroll
                for (int i = 0; i < 8; ++i)
                    bv.u[i] = *reinterpret_cast<const unsigned*>(
                        &VtT[cur][ss * 16 + l16][a_kbase(i, half)]);
                o[ss].v = __builtin_amdgcn_wmma_f32_16x16x32_bf16(
                    false, pf.v, false, bv.v, (short)0, o[ss].v, false, false);
            }
        }

        if (more) {                                // stage next V tile (other buffer)
            const int nxt = cur ^ 1;
            #pragma unroll
            for (int i = 0; i < 8; ++i) VtT[nxt][d0 + i][kk] = vv.h[i];
        }
    }
    // normalize and write y (bf16)
    #pragma unroll
    for (int r = 0; r < 8; ++r) {
        float inv = 1.0f / l_run[r];
        size_t row = (size_t)(b * GPT_T + qr0 + r + 8 * half);
        bf16* yrow = y + row * GPT_D + h * 64;
        #pragma unroll
        for (int ss = 0; ss < 4; ++ss)
            yrow[ss * 16 + l16] = (bf16)(o[ss].f[r] * inv);
    }
}

// ---------------------------------------------------------------- logits (2 x V)
// wave-per-row dot products; coalesced sweep of wte (bandwidth bound).
__global__ __launch_bounds__(256) void logits_kernel(const float* __restrict__ xr2,
                                                     const float* __restrict__ wte,
                                                     float* __restrict__ out) {
    __shared__ float xs[GPT_D];
    const int b = blockIdx.y;
    for (int i = threadIdx.x; i < GPT_D; i += 256) xs[i] = xr2[b * GPT_D + i];
    __syncthreads();
    const int lane = threadIdx.x & 31, wid = threadIdx.x >> 5;
    const int v = blockIdx.x * 8 + wid;
    if (v >= GPT_V) return;
    const float* wr = wte + (size_t)v * GPT_D;
    float acc = 0.f;
    for (int i = lane; i < GPT_D; i += 32) acc += wr[i] * xs[i];
    #pragma unroll
    for (int o = 16; o > 0; o >>= 1) acc += __shfl_xor(acc, o, 32);
    if (lane == 0) out[(size_t)b * GPT_V + v] = acc;
}

// ----------------------------------------------------------------------------
extern "C" void kernel_launch(void* const* d_in, const int* in_sizes, int n_in,
                              void* d_out, int out_size, void* d_ws, size_t ws_size,
                              hipStream_t stream) {
    (void)in_sizes; (void)n_in; (void)out_size; (void)ws_size;

    const int*   idx    = (const int*)  d_in[0];
    const float* wte    = (const float*)d_in[1];
    const float* wpe    = (const float*)d_in[2];
    const float* ln1_w  = (const float*)d_in[3];
    const float* ln1_b  = (const float*)d_in[4];
    const float* attn_w = (const float*)d_in[5];
    const float* attn_b = (const float*)d_in[6];
    const float* proj_w = (const float*)d_in[7];
    const float* proj_b = (const float*)d_in[8];
    const float* ln2_w  = (const float*)d_in[9];
    const float* ln2_b  = (const float*)d_in[10];
    const float* fc_w   = (const float*)d_in[11];
    const float* fc_b   = (const float*)d_in[12];
    const float* fcp_w  = (const float*)d_in[13];
    const float* fcp_b  = (const float*)d_in[14];
    const float* lnf_w  = (const float*)d_in[15];
    const float* lnf_b  = (const float*)d_in[16];

    char* ws = (char*)d_ws;
    auto al = [](size_t x) { return (x + 255) & ~(size_t)255; };
    size_t off = 0;
    size_t OFF_X    = off; off += al((size_t)GPT_M * GPT_D * 4);
    size_t OFF_H    = off; off += al((size_t)GPT_M * GPT_D * 2);
    size_t OFF_QKV  = off; off += al((size_t)GPT_M * GPT_3D * 2);
    size_t OFF_Y    = off; off += al((size_t)GPT_M * GPT_D * 2);
    size_t OFF_MACT = off; off += al((size_t)GPT_M * GPT_4D * 2);
    size_t OFF_AW   = off; off += al((size_t)GPT_L * GPT_D * GPT_3D * 2);
    size_t OFF_PW   = off; off += al((size_t)GPT_L * GPT_D * GPT_D * 2);
    size_t OFF_FW   = off; off += al((size_t)GPT_L * GPT_D * GPT_4D * 2);
    size_t OFF_FPW  = off; off += al((size_t)GPT_L * GPT_4D * GPT_D * 2);
    size_t OFF_LNF  = off; off += al((size_t)2 * GPT_D * 4);

    float* X    = (float*)(ws + OFF_X);
    bf16*  Hb   = (bf16*) (ws + OFF_H);
    bf16*  QKVb = (bf16*) (ws + OFF_QKV);
    bf16*  Yb   = (bf16*) (ws + OFF_Y);
    bf16*  Mb   = (bf16*) (ws + OFF_MACT);
    bf16*  AW   = (bf16*) (ws + OFF_AW);     // [L, 3D, D]   (N-major, transposed)
    bf16*  PW   = (bf16*) (ws + OFF_PW);     // [L, D, D]
    bf16*  FW   = (bf16*) (ws + OFF_FW);     // [L, 4D, D]
    bf16*  FPW  = (bf16*) (ws + OFF_FPW);    // [L, D, 4D]
    float* LNF  = (float*)(ws + OFF_LNF);

    // weights fp32 [K,N] -> bf16 [N,K] (once per launch)
    cvt_transpose_kernel<<<dim3(GPT_3D / 64, GPT_D / 64, GPT_L), 256, 0, stream>>>(
        attn_w, AW, GPT_D, GPT_3D);
    cvt_transpose_kernel<<<dim3(GPT_D / 64, GPT_D / 64, GPT_L), 256, 0, stream>>>(
        proj_w, PW, GPT_D, GPT_D);
    cvt_transpose_kernel<<<dim3(GPT_4D / 64, GPT_D / 64, GPT_L), 256, 0, stream>>>(
        fc_w, FW, GPT_D, GPT_4D);
    cvt_transpose_kernel<<<dim3(GPT_D / 64, GPT_4D / 64, GPT_L), 256, 0, stream>>>(
        fcp_w, FPW, GPT_4D, GPT_D);

    embed_kernel<<<GPT_M, 256, 0, stream>>>(idx, wte, wpe, X);

    for (int l = 0; l < GPT_L; ++l) {
        ln_kernel<<<GPT_M / 8, 256, 0, stream>>>(X, ln1_w + l * GPT_D, ln1_b + l * GPT_D, Hb);
        gemm_kernel<<<dim3(GPT_3D / 64, GPT_M / 128), 256, 0, stream>>>(
            Hb, AW + (size_t)l * GPT_D * GPT_3D, attn_b + l * GPT_3D,
            nullptr, nullptr, QKVb, GPT_M, GPT_3D, GPT_D, 0);
        attn_kernel<<<dim3(GPT_T / 128, 12, 2), 256, 0, stream>>>(QKVb, Yb);
        gemm_kernel<<<dim3(GPT_D / 64, GPT_M / 128), 256, 0, stream>>>(
            Yb, PW + (size_t)l * GPT_D * GPT_D, proj_b + l * GPT_D,
            X, X, nullptr, GPT_M, GPT_D, GPT_D, 0);
        ln_kernel<<<GPT_M / 8, 256, 0, stream>>>(X, ln2_w + l * GPT_D, ln2_b + l * GPT_D, Hb);
        gemm_kernel<<<dim3(GPT_4D / 64, GPT_M / 128), 256, 0, stream>>>(
            Hb, FW + (size_t)l * GPT_D * GPT_4D, fc_b + l * GPT_4D,
            nullptr, nullptr, Mb, GPT_M, GPT_4D, GPT_D, 1);
        gemm_kernel<<<dim3(GPT_D / 64, GPT_M / 128), 256, 0, stream>>>(
            Mb, FPW + (size_t)l * GPT_4D * GPT_D, fcp_b + l * GPT_D,
            X, X, nullptr, GPT_M, GPT_D, GPT_4D, 0);
    }

    lnf_kernel<<<1, 64, 0, stream>>>(X, lnf_w, lnf_b, LNF);
    logits_kernel<<<dim3((GPT_V + 7) / 8, 2), 256, 0, stream>>>(LNF, wte, (float*)d_out);
}
